// DecoderBlock_11613591568704
// MI455X (gfx1250) — matrix operand order repaired and checked by reading, not compiled, not run on hardware
//
#include <hip/hip_runtime.h>
#include <hip/hip_bf16.h>

// ---------------------------------------------------------------------------
// Decoder block (linear attention) for MI455X / gfx1250, wave32 + WMMA bf16.
// GEMM: 32x32 register-blocked tile per wave (4 WMMAs / k-step, 2 A + 2 B frags).
// ---------------------------------------------------------------------------

typedef __bf16 bf16;
typedef __attribute__((ext_vector_type(16))) __bf16 v16bf;
typedef __attribute__((ext_vector_type(8)))  float  v8f;

#define WMMA_BF16(a, b, c) \
  __builtin_amdgcn_wmma_f32_16x16x32_bf16(false, (a), false, (b), (short)0, (c), false, false)

static constexpr int BB  = 2;     // batch
static constexpr int TT  = 1024;  // target seq
static constexpr int SS  = 1024;  // memory seq
static constexpr int CC  = 1024;  // channels
static constexpr int HH  = 16;    // heads
static constexpr int HD  = 64;    // head dim
static constexpr float LN_EPS = 1e-5f;

__device__ __forceinline__ bf16 tobf(float f) { return (bf16)f; }

__device__ __forceinline__ float gelu_tanh(float x) {
  const float k = 0.7978845608028654f;  // sqrt(2/pi)
  float x3 = x * x * x;
  return 0.5f * x * (1.f + tanhf(k * (x + 0.044715f * x3)));
}

// ---- WMMA fragment loaders (ISA 7.12.2 layouts, wave32) --------------------
// A: 16(M) x 32(K) bf16, row-major with leading dim ld.
//   lane m = lane&15; elems 0..7 = K{0..7|8..15}, elems 8..15 = K{16..23|24..31}
__device__ __forceinline__ v16bf load_frag_a(const bf16* base, int ld, int lane) {
  const bf16* p = base + (size_t)(lane & 15) * ld + ((lane & 16) ? 8 : 0);
  union { v16bf v; float4 f[2]; } u;
  u.f[0] = *(const float4*)(p);
  u.f[1] = *(const float4*)(p + 16);
  return u.v;
}
// B: 32(K) x 16(N) bf16, stored transposed as [N][K] with leading dim ld.
//   lane n = lane&15; elems 0..7 = K{0..7|8..15}, elems 8..15 = K{16..23|24..31}
__device__ __forceinline__ v16bf load_frag_b(const bf16* base, int ld, int lane) {
  const bf16* p = base + (size_t)(lane & 15) * ld + ((lane & 16) ? 8 : 0);
  union { v16bf v; float4 f[2]; } u;
  u.f[0] = *(const float4*)(p);
  u.f[1] = *(const float4*)(p + 16);
  return u.v;
}

// ---------------------------------------------------------------------------
// Weight convert + transpose: Wt[n*K + k] = (bf16) W[k*N + n]
// ---------------------------------------------------------------------------
__global__ void k_convert_transpose(const float* __restrict__ W, bf16* __restrict__ Wt,
                                    int K, int N) {
  int n = blockIdx.y;
  int k = blockIdx.x * 256 + threadIdx.x;
  if (k < K) Wt[(size_t)n * K + k] = tobf(W[(size_t)k * N + n]);
}

__global__ void k_convert(const float* __restrict__ X, bf16* __restrict__ Y, int n) {
  int i = blockIdx.x * 256 + threadIdx.x;
  if (i < n) Y[i] = tobf(X[i]);
}

// ---------------------------------------------------------------------------
// LayerNorm over last dim C (one block per row), bf16 output.
// ---------------------------------------------------------------------------
__global__ void k_layernorm(const float* __restrict__ x, const float* __restrict__ g,
                            const float* __restrict__ b, bf16* __restrict__ out, int C) {
  __shared__ float red[256];
  int row = blockIdx.x, tid = threadIdx.x;
  const float* xr = x + (size_t)row * C;

  float loc = 0.f;
  for (int i = tid; i < C; i += 256) loc += xr[i];
  red[tid] = loc; __syncthreads();
  for (int s = 128; s > 0; s >>= 1) { if (tid < s) red[tid] += red[tid + s]; __syncthreads(); }
  float mu = red[0] / C; __syncthreads();

  loc = 0.f;
  for (int i = tid; i < C; i += 256) { float d = xr[i] - mu; loc += d * d; }
  red[tid] = loc; __syncthreads();
  for (int s = 128; s > 0; s >>= 1) { if (tid < s) red[tid] += red[tid + s]; __syncthreads(); }
  float rstd = rsqrtf(red[0] / C + LN_EPS);

  bf16* orow = out + (size_t)row * C;
  for (int i = tid; i < C; i += 256)
    orow[i] = tobf((xr[i] - mu) * rstd * g[i] + b[i]);
}

// ---------------------------------------------------------------------------
// GEMM: C[M,N] = A[M,K](bf16) @ Bt[N,K](bf16)^T + bias (+ residual) (+ GELU)
// 32x32 tile per wave: 2 A-frags x 2 B-frags -> 4 WMMAs per 32-wide k-step.
// 4 waves/block -> block tile = 32 rows x 128 cols.
// ---------------------------------------------------------------------------
__global__ __launch_bounds__(128)
void k_gemm(const bf16* __restrict__ A, const bf16* __restrict__ Bt,
            const float* __restrict__ bias, const float* __restrict__ res,
            float* __restrict__ Cf, bf16* __restrict__ Cb,
            int M, int N, int K, int dogelu) {
  int lane = threadIdx.x & 31;
  int wave = threadIdx.x >> 5;
  int n0 = blockIdx.x * 128 + wave * 32;
  int m0 = blockIdx.y * 32;
  if (n0 >= N || m0 >= M) return;

  const bf16* Arow0 = A  + (size_t)m0 * K;
  const bf16* Arow1 = A  + (size_t)(m0 + 16) * K;
  const bf16* Brow0 = Bt + (size_t)n0 * K;
  const bf16* Brow1 = Bt + (size_t)(n0 + 16) * K;
  v8f acc00 = {}, acc01 = {}, acc10 = {}, acc11 = {};
  for (int k0 = 0; k0 < K; k0 += 32) {
    v16bf a0 = load_frag_a(Arow0 + k0, K, lane);
    v16bf a1 = load_frag_a(Arow1 + k0, K, lane);
    v16bf b0 = load_frag_b(Brow0 + k0, K, lane);
    v16bf b1 = load_frag_b(Brow1 + k0, K, lane);
    acc00 = WMMA_BF16(a0, b0, acc00);
    acc01 = WMMA_BF16(a0, b1, acc01);
    acc10 = WMMA_BF16(a1, b0, acc10);
    acc11 = WMMA_BF16(a1, b1, acc11);
  }

  int nl = lane & 15;
  int mb = (lane & 16) ? 8 : 0;
  v8f accs[2][2] = {{acc00, acc01}, {acc10, acc11}};
#pragma unroll
  for (int jm = 0; jm < 2; ++jm) {
#pragma unroll
    for (int jn = 0; jn < 2; ++jn) {
      int n = n0 + jn * 16 + nl;
      float bv = bias ? bias[n] : 0.f;
#pragma unroll
      for (int i = 0; i < 8; ++i) {
        int m = m0 + jm * 16 + mb + i;
        float v = accs[jm][jn][i] + bv;
        if (res)    v += res[(size_t)m * N + n];
        if (dogelu) v  = gelu_tanh(v);
        if (Cf) Cf[(size_t)m * N + n] = v;
        if (Cb) Cb[(size_t)m * N + n] = tobf(v);
      }
    }
  }
}

// ---------------------------------------------------------------------------
// Featurize: ELU(x)+1 (== x>0 ? x+1 : exp(x)), then RoPE.  Per-head layouts:
//   Qf fp32 [B,H,T,HD], Qr bf16 [B,H,T,HD]; K likewise; Vt bf16 [B,H,HD,S].
// ---------------------------------------------------------------------------
__global__ void k_feat_q(const float* __restrict__ src, int stride, int off,
                         float* __restrict__ Qf, bf16* __restrict__ Qr) {
  __shared__ float sh[64];
  int bt = blockIdx.x;            // b*T + t
  int t  = bt % TT, b = bt / TT;
  int d  = threadIdx.x;           // 0..63
  float inv = expf(-(float)(d & 31) * 0.28782313662425575f);  // ln(1e4)/32
  float ang = (float)t * inv;
  float c = cosf(ang), s = sinf(ang);
  for (int h = 0; h < HH; ++h) {
    float x = src[(size_t)bt * stride + off + h * HD + d];
    float f = x > 0.f ? x + 1.f : expf(x);
    sh[d] = f; __syncthreads();
    float part = sh[d ^ 32]; __syncthreads();
    float rot = (d < 32) ? -part : part;
    size_t o = ((size_t)(b * HH + h) * TT + t) * HD + d;
    Qf[o] = f;
    Qr[o] = tobf(f * c + rot * s);
  }
}

__global__ void k_feat_kv(const float* __restrict__ src, int stride, int koff, int voff,
                          float* __restrict__ Kf, bf16* __restrict__ Kr,
                          bf16* __restrict__ Vt) {
  __shared__ float sh[64];
  int bs = blockIdx.x;            // b*S + s
  int t  = bs % SS, b = bs / SS;
  int d  = threadIdx.x;
  float inv = expf(-(float)(d & 31) * 0.28782313662425575f);
  float ang = (float)t * inv;
  float c = cosf(ang), s = sinf(ang);
  for (int h = 0; h < HH; ++h) {
    float x = src[(size_t)bs * stride + koff + h * HD + d];
    float f = x > 0.f ? x + 1.f : expf(x);
    sh[d] = f; __syncthreads();
    float part = sh[d ^ 32]; __syncthreads();
    float rot = (d < 32) ? -part : part;
    size_t o = ((size_t)(b * HH + h) * SS + t) * HD + d;
    Kf[o] = f;
    Kr[o] = tobf(f * c + rot * s);
    float v = src[(size_t)bs * stride + voff + h * HD + d];
    Vt[((size_t)(b * HH + h) * HD + d) * SS + t] = tobf(v);
  }
}

// Ksum[bh][d] = sum_s Kf[bh][s][d]   (deterministic serial reduction)
__global__ void k_ksum(const float* __restrict__ Kf, float* __restrict__ Ksum) {
  int bh = blockIdx.x, d = threadIdx.x;
  float s = 0.f;
  for (int t = 0; t < SS; ++t) s += Kf[((size_t)bh * SS + t) * HD + d];
  Ksum[(size_t)bh * HD + d] = s;
}

// invden[bh][t] = 1 / (Qf[bh][t][:] . Ksum[bh][:])
__global__ void k_denom(const float* __restrict__ Qf, const float* __restrict__ Ksum,
                        float* __restrict__ invden) {
  int idx = blockIdx.x * 256 + threadIdx.x;   // < B*H*T
  if (idx >= BB * HH * TT) return;
  int bh = idx / TT, t = idx % TT;
  float s = 0.f;
  for (int d = 0; d < HD; ++d)
    s += Qf[((size_t)bh * TT + t) * HD + d] * Ksum[(size_t)bh * HD + d];
  invden[idx] = 1.f / s;
}

// ---------------------------------------------------------------------------
// Flash-style linear attention.  One wave owns 16 query rows; loops over S in
// chunks of 32: score tiles (WMMA over d), causal mask in f32 fragment, LDS
// round-trip to re-fragment as A, then y += att @ V (WMMA over s, Vt B-frags).
// ---------------------------------------------------------------------------
__global__ __launch_bounds__(128)
void k_attn(const bf16* __restrict__ Qr, const bf16* __restrict__ Kr,
            const bf16* __restrict__ Vt, const float* __restrict__ invden,
            bf16* __restrict__ Y, int causal) {
  __shared__ __attribute__((aligned(16))) bf16 latt[4][16][32];
  int lane = threadIdx.x & 31;
  int wave = threadIdx.x >> 5;
  int bh = blockIdx.y;
  int t0 = blockIdx.x * 64 + wave * 16;

  const bf16* Qbase = Qr + ((size_t)bh * TT + t0) * HD;  // rows t, ld = HD
  const bf16* Kbase = Kr + (size_t)bh * SS * HD;          // rows s, ld = HD
  const bf16* Vbase = Vt + (size_t)bh * HD * SS;          // rows d, ld = SS

  v8f y0 = {}, y1 = {}, y2 = {}, y3 = {};
  int n  = lane & 15;
  int mb = (lane & 16) ? 8 : 0;
  int send = causal ? (t0 + 16) : SS;

  for (int s0 = 0; s0 < send; s0 += 32) {
    // scores: 16(t) x 32(s), contraction over d = 64
    v8f sc0 = {}, sc1 = {};
    for (int k0 = 0; k0 < HD; k0 += 32) {
      v16bf a  = load_frag_a(Qbase + k0, HD, lane);
      v16bf b0 = load_frag_b(Kbase + (size_t)s0 * HD + k0, HD, lane);
      v16bf b1 = load_frag_b(Kbase + (size_t)(s0 + 16) * HD + k0, HD, lane);
      sc0 = WMMA_BF16(a, b0, sc0);
      sc1 = WMMA_BF16(a, b1, sc1);
    }
    // causal mask + round-trip through LDS as a bf16 A-tile [16][32]
#pragma unroll
    for (int i = 0; i < 8; ++i) {
      int t = t0 + mb + i;
      float v0 = sc0[i]; if (causal && (s0 + n)      > t) v0 = 0.f;
      float v1 = sc1[i]; if (causal && (s0 + 16 + n) > t) v1 = 0.f;
      latt[wave][mb + i][n]      = tobf(v0);
      latt[wave][mb + i][16 + n] = tobf(v1);
    }
    v16bf aat = load_frag_a(&latt[wave][0][0], 32, lane);
    y0 = WMMA_BF16(aat, load_frag_b(Vbase + (size_t)0 * 16 * SS + s0, SS, lane), y0);
    y1 = WMMA_BF16(aat, load_frag_b(Vbase + (size_t)1 * 16 * SS + s0, SS, lane), y1);
    y2 = WMMA_BF16(aat, load_frag_b(Vbase + (size_t)2 * 16 * SS + s0, SS, lane), y2);
    y3 = WMMA_BF16(aat, load_frag_b(Vbase + (size_t)3 * 16 * SS + s0, SS, lane), y3);
  }

  // store: y[b*T+t, h*64 + d] * (1/denom[t]); division is row-linear so defer here
  int b = bh / HH, h = bh % HH;
#pragma unroll
  for (int i = 0; i < 8; ++i) {
    int t = t0 + mb + i;
    float id = invden[(size_t)bh * TT + t];
    bf16* yr = Y + (size_t)(b * TT + t) * CC + h * HD;
    yr[0 * 16 + n] = tobf(y0[i] * id);
    yr[1 * 16 + n] = tobf(y1[i] * id);
    yr[2 * 16 + n] = tobf(y2[i] * id);
    yr[3 * 16 + n] = tobf(y3[i] * id);
  }
}

// ---------------------------------------------------------------------------
// Host orchestration
// ---------------------------------------------------------------------------
extern "C" void kernel_launch(void* const* d_in, const int* in_sizes, int n_in,
                              void* d_out, int out_size, void* d_ws, size_t ws_size,
                              hipStream_t stream) {
  (void)in_sizes; (void)n_in; (void)out_size; (void)ws_size;
  const float* x        = (const float*)d_in[0];
  const float* memory   = (const float*)d_in[1];
  const float* ln1_g    = (const float*)d_in[2];
  const float* ln1_b    = (const float*)d_in[3];
  const float* sa_qkv_w = (const float*)d_in[4];
  const float* sa_qkv_b = (const float*)d_in[5];
  const float* sa_pj_w  = (const float*)d_in[6];
  const float* sa_pj_b  = (const float*)d_in[7];
  const float* ln2_g    = (const float*)d_in[8];
  const float* ln2_b    = (const float*)d_in[9];
  const float* ca_q_w   = (const float*)d_in[10];
  const float* ca_q_b   = (const float*)d_in[11];
  const float* ca_kv_w  = (const float*)d_in[12];
  const float* ca_kv_b  = (const float*)d_in[13];
  const float* ca_pj_w  = (const float*)d_in[14];
  const float* ca_pj_b  = (const float*)d_in[15];
  const float* ln3_g    = (const float*)d_in[16];
  const float* ln3_b    = (const float*)d_in[17];
  const float* fc_w     = (const float*)d_in[18];
  const float* fc_b     = (const float*)d_in[19];
  const float* fcp_w    = (const float*)d_in[20];
  const float* fcp_b    = (const float*)d_in[21];

  const int M  = BB * TT;   // 2048
  const int BH = BB * HH;   // 32

  // bump allocator on d_ws (256B aligned)
  char* base = (char*)d_ws; size_t off = 0;
  auto alloc = [&](size_t bytes) -> void* {
    off = (off + 255) & ~(size_t)255;
    void* p = base + off; off += bytes; return p;
  };
  bf16*  WqkvT = (bf16*)alloc((size_t)3*CC*CC*2);
  bf16*  WsapT = (bf16*)alloc((size_t)CC*CC*2);
  bf16*  WcaqT = (bf16*)alloc((size_t)CC*CC*2);
  bf16*  WcakT = (bf16*)alloc((size_t)2*CC*CC*2);
  bf16*  WcapT = (bf16*)alloc((size_t)CC*CC*2);
  bf16*  WfcT  = (bf16*)alloc((size_t)4*CC*CC*2);
  bf16*  WfcpT = (bf16*)alloc((size_t)4*CC*CC*2);
  bf16*  mem_bf= (bf16*)alloc((size_t)BB*SS*CC*2);
  bf16*  h_ln  = (bf16*)alloc((size_t)M*CC*2);
  bf16*  h_fc  = (bf16*)alloc((size_t)M*4*CC*2);
  float* qkv   = (float*)alloc((size_t)M*3*CC*4);
  float* q_ca  = (float*)alloc((size_t)M*CC*4);
  float* kv_ca = (float*)alloc((size_t)BB*SS*2*CC*4);
  float* Qf    = (float*)alloc((size_t)BH*TT*HD*4);
  float* Kf    = (float*)alloc((size_t)BH*SS*HD*4);
  bf16*  Qr    = (bf16*)alloc((size_t)BH*TT*HD*2);
  bf16*  Kr    = (bf16*)alloc((size_t)BH*SS*HD*2);
  bf16*  Vt    = (bf16*)alloc((size_t)BH*HD*SS*2);
  float* Ksum  = (float*)alloc((size_t)BH*HD*4);
  float* invden= (float*)alloc((size_t)BH*TT*4);
  bf16*  y_att = (bf16*)alloc((size_t)M*CC*2);
  float* x1    = (float*)alloc((size_t)M*CC*4);
  float* x2    = (float*)alloc((size_t)M*CC*4);

  auto xpose = [&](const float* W, bf16* Wt, int K, int N) {
    k_convert_transpose<<<dim3(K/256, N), 256, 0, stream>>>(W, Wt, K, N);
  };
  auto gemm = [&](const bf16* A, const bf16* Bt, const float* bias, const float* res,
                  float* Cf, bf16* Cb, int Mm, int Nn, int Kk, int dogelu) {
    k_gemm<<<dim3(Nn/128, Mm/32), 128, 0, stream>>>(A, Bt, bias, res, Cf, Cb, Mm, Nn, Kk, dogelu);
  };

  // weight conversion (transposed bf16)
  xpose(sa_qkv_w, WqkvT, CC, 3*CC);
  xpose(sa_pj_w,  WsapT, CC, CC);
  xpose(ca_q_w,   WcaqT, CC, CC);
  xpose(ca_kv_w,  WcakT, CC, 2*CC);
  xpose(ca_pj_w,  WcapT, CC, CC);
  xpose(fc_w,     WfcT,  CC, 4*CC);
  xpose(fcp_w,    WfcpT, 4*CC, CC);
  k_convert<<<(BB*SS*CC + 255)/256, 256, 0, stream>>>(memory, mem_bf, BB*SS*CC);

  // ---- causal self-attention ----
  k_layernorm<<<M, 256, 0, stream>>>(x, ln1_g, ln1_b, h_ln, CC);
  gemm(h_ln, WqkvT, sa_qkv_b, nullptr, qkv, nullptr, M, 3*CC, CC, 0);
  k_feat_q <<<M, 64, 0, stream>>>(qkv, 3*CC, 0, Qf, Qr);
  k_feat_kv<<<M, 64, 0, stream>>>(qkv, 3*CC, CC, 2*CC, Kf, Kr, Vt);
  k_ksum <<<BH, 64, 0, stream>>>(Kf, Ksum);
  k_denom<<<(BH*TT + 255)/256, 256, 0, stream>>>(Qf, Ksum, invden);
  k_attn <<<dim3(TT/64, BH), 128, 0, stream>>>(Qr, Kr, Vt, invden, y_att, 1);
  gemm(y_att, WsapT, sa_pj_b, x, x1, nullptr, M, CC, CC, 0);

  // ---- cross-attention ----
  k_layernorm<<<M, 256, 0, stream>>>(x1, ln2_g, ln2_b, h_ln, CC);
  gemm(h_ln,   WcaqT, ca_q_b,  nullptr, q_ca,  nullptr, M, CC,   CC, 0);
  gemm(mem_bf, WcakT, ca_kv_b, nullptr, kv_ca, nullptr, BB*SS, 2*CC, CC, 0);
  k_feat_q <<<M, 64, 0, stream>>>(q_ca, CC, 0, Qf, Qr);
  k_feat_kv<<<BB*SS, 64, 0, stream>>>(kv_ca, 2*CC, 0, CC, Kf, Kr, Vt);
  k_ksum <<<BH, 64, 0, stream>>>(Kf, Ksum);
  k_denom<<<(BH*TT + 255)/256, 256, 0, stream>>>(Qf, Ksum, invden);
  k_attn <<<dim3(TT/64, BH), 128, 0, stream>>>(Qr, Kr, Vt, invden, y_att, 0);
  gemm(y_att, WcapT, ca_pj_b, x1, x2, nullptr, M, CC, CC, 0);

  // ---- MLP with tanh GELU ----
  k_layernorm<<<M, 256, 0, stream>>>(x2, ln3_g, ln3_b, h_ln, CC);
  gemm(h_ln, WfcT, fc_b, nullptr, nullptr, h_fc, M, 4*CC, CC, 1);
  gemm(h_fc, WfcpT, fcp_b, x2, (float*)d_out, nullptr, M, CC, 4*CC, 0);
}
